// NgramHashEmbeddingSample_37812892074113
// MI455X (gfx1250) — compile-verified, matrix operand order checked by reading, not run on hardware
//
#include <hip/hip_runtime.h>

typedef __attribute__((ext_vector_type(2)))  float    v2f;
typedef __attribute__((ext_vector_type(8)))  float    v8f;
typedef __attribute__((ext_vector_type(4)))  _Float16 v4h;
typedef __attribute__((ext_vector_type(8)))  _Float16 v8h;
typedef __attribute__((ext_vector_type(16))) _Float16 v16h;

#define S_LEN   2048
#define EMB     64
#define KDIM    1024      // 16 tables * 64
#define NOUT    1024
#define BM      64        // rows per block
#define BN      256       // output columns per block
#define KH      512       // K half (8 tables)
#define NPAIR   (BM * 8)  // (row, table) pairs per K-half
#define XSTR_H  520       // f16 LDS row stride in halves (512 + 8 pad)
#define XSTRIDE 516       // f32 fallback LDS stride in floats
#define NT16    (NOUT / 16)
#define LSCALE     2048.0f
#define INV_LSCALE (1.0f / 2048.0f)

// ---------------- shared hash helper ----------------
__device__ __forceinline__ unsigned hash_index(
    const int* __restrict__ tok, const int* __restrict__ mults,
    const int* __restrict__ bias, const int* __restrict__ sizes,
    const int* __restrict__ offs, int gr, int t)
{
    const int b = gr >> 11;            // / 2048
    const int s = gr & (S_LEN - 1);
    const int* tp = tok + b * S_LEN + s;
    const unsigned long long t0 = (unsigned)tp[0];
    const unsigned long long t1 = (s >= 1) ? (unsigned)tp[-1] : 0ull;
    const unsigned long long m0 = (unsigned)mults[t * 3 + 0];
    const unsigned long long m1 = (unsigned)mults[t * 3 + 1];
    unsigned long long hsh = (m0 * t0) ^ (m1 * t1);
    if (t >= 8) {                      // order-3 tables include k=2 term
        const unsigned long long t2 = (s >= 2) ? (unsigned)tp[-2] : 0ull;
        const unsigned long long m2 = (unsigned)mults[t * 3 + 2];
        hsh ^= m2 * t2;
    }
    hsh ^= (unsigned long long)(unsigned)bias[t];
    return (unsigned)(hsh % (unsigned long long)(unsigned)sizes[t]) + (unsigned)offs[t];
}

// ---------------- pack W into fragment-native hi/lo f16 ----------------
// Block (kb, nt) = 256 dwords: dword = {K0, K0+1} f16 pair for col n(lane).
// K0 = 2*(v&3) + 16*(v>>2) + 8*laneH ; fragment load = 2 x contiguous 512B.
__launch_bounds__(256)
__global__ void pack_w_f16(const float* __restrict__ W,
                           unsigned* __restrict__ WhP,
                           unsigned* __restrict__ WlP)
{
    const int idx = blockIdx.x * 256 + threadIdx.x;   // 0 .. 1024*512-1
    const int n   = idx >> 9;
    const int k   = (idx & 511) << 1;
    const float w0 = W[(size_t)n * KDIM + k];
    const float w1 = W[(size_t)n * KDIM + k + 1];
    const _Float16 h0 = (_Float16)w0, h1 = (_Float16)w1;
    const _Float16 l0 = (_Float16)((w0 - (float)h0) * LSCALE);
    const _Float16 l1 = (_Float16)((w1 - (float)h1) * LSCALE);
    const int kb = k >> 5, kk = k & 31, nt = n >> 4, nl = n & 15;
    const int v    = ((kk >> 1) & 3) | (((kk >> 4) & 1) << 2);
    const int lane = nl | (((kk >> 3) & 1) << 4);
    const int dw   = ((kb * NT16 + nt) << 8) + ((v & 4) ? 128 : 0) + lane * 4 + (v & 3);
    const unsigned hu = ((unsigned)__builtin_bit_cast(unsigned short, h1) << 16)
                      |  (unsigned)__builtin_bit_cast(unsigned short, h0);
    const unsigned lu = ((unsigned)__builtin_bit_cast(unsigned short, l1) << 16)
                      |  (unsigned)__builtin_bit_cast(unsigned short, l0);
    WhP[dw] = hu;
    WlP[dw] = lu;
}

#define SHUF16(a, b) __builtin_shufflevector(a, b, 0,1,2,3,4,5,6,7,8,9,10,11,12,13,14,15)

// ---------------- fused hash + gather + split-f16 WMMA GEMM ----------------
__launch_bounds__(256)
__global__ void ngram_gemm_f16split(const int*      __restrict__ tok,
                                    const float*    __restrict__ table,
                                    const unsigned* __restrict__ WhP,
                                    const unsigned* __restrict__ WlP,
                                    const int* __restrict__ mults,
                                    const int* __restrict__ bias,
                                    const int* __restrict__ sizes,
                                    const int* __restrict__ offs,
                                    float* __restrict__ out)
{
    __shared__ _Float16 Xh[BM * XSTR_H];
    __shared__ _Float16 Xl[BM * XSTR_H];
    __shared__ unsigned s_idx[NPAIR];

    const int tid   = threadIdx.x;
    const int lane  = tid & 31;
    const int wave  = tid >> 5;
    const int laneM = lane & 15;
    const int laneH = lane >> 4;
    const int rowBase  = blockIdx.x * BM;
    const int ctBase   = blockIdx.y * (BN / 16);   // first col tile of block

    v8f acc1[4][2], acc2[4][2];
    const v8f vz = {0.f, 0.f, 0.f, 0.f, 0.f, 0.f, 0.f, 0.f};
#pragma unroll
    for (int rt = 0; rt < 4; ++rt)
#pragma unroll
        for (int cj = 0; cj < 2; ++cj) { acc1[rt][cj] = vz; acc2[rt][cj] = vz; }

    for (int h = 0; h < 2; ++h) {
        __syncthreads();
        // ---- hash: 512 pairs, 2 per thread ----
#pragma unroll
        for (int q = 0; q < 2; ++q) {
            const int p = q * 256 + tid;
            s_idx[p] = hash_index(tok, mults, bias, sizes, offs,
                                  rowBase + (p >> 3), h * 8 + (p & 7));
        }
        __syncthreads();
        // ---- gather + hi/lo f16 split into LDS ----
#pragma unroll 4
        for (int i = 0; i < 32; ++i) {
            const int c  = i * 256 + tid;
            const int p  = c >> 4;
            const int e4 = c & 15;
            const int r  = p >> 3;
            const int tt = p & 7;
            const float4 vv = *(const float4*)(table + (size_t)s_idx[p] * EMB + e4 * 4);
            const _Float16 a = (_Float16)vv.x, b = (_Float16)vv.y,
                           cc = (_Float16)vv.z, d = (_Float16)vv.w;
            const v4h hi = {a, b, cc, d};
            const v4h lo = {(_Float16)((vv.x - (float)a)  * LSCALE),
                            (_Float16)((vv.y - (float)b)  * LSCALE),
                            (_Float16)((vv.z - (float)cc) * LSCALE),
                            (_Float16)((vv.w - (float)d)  * LSCALE)};
            const int off = r * XSTR_H + tt * EMB + e4 * 4;
            *(v4h*)&Xh[off] = hi;
            *(v4h*)&Xl[off] = lo;
        }
        __syncthreads();

        // ---- GEMM: 16 K-blocks of 32 ----
        for (int kb = 0; kb < 16; ++kb) {
            const int kg = h * 16 + kb;
            v16h Bh[2], Bl[2];
#pragma unroll
            for (int cj = 0; cj < 2; ++cj) {
                const int ct = ctBase + wave * 2 + cj;
                const unsigned* blkH = WhP + ((kg * NT16 + ct) << 8);
                const unsigned* blkL = WlP + ((kg * NT16 + ct) << 8);
                const v8h bh0 = *(const v8h*)(blkH + lane * 4);
                const v8h bh1 = *(const v8h*)(blkH + 128 + lane * 4);
                const v8h bl0 = *(const v8h*)(blkL + lane * 4);
                const v8h bl1 = *(const v8h*)(blkL + 128 + lane * 4);
                Bh[cj] = SHUF16(bh0, bh1);
                Bl[cj] = SHUF16(bl0, bl1);
            }
#pragma unroll
            for (int rt = 0; rt < 4; ++rt) {
                const int offA = (rt * 16 + laneM) * XSTR_H + kb * 32 + 8 * laneH;
                const v8h ah0 = *(const v8h*)&Xh[offA];
                const v8h ah1 = *(const v8h*)&Xh[offA + 16];
                const v8h al0 = *(const v8h*)&Xl[offA];
                const v8h al1 = *(const v8h*)&Xl[offA + 16];
                const v16h Ah = SHUF16(ah0, ah1);
                const v16h Al = SHUF16(al0, al1);
#pragma unroll
                for (int cj = 0; cj < 2; ++cj) {
                    acc1[rt][cj] = __builtin_amdgcn_wmma_f32_16x16x32_f16(
                        false, Ah, false, Bh[cj], (short)0, acc1[rt][cj], false, false);
                    acc2[rt][cj] = __builtin_amdgcn_wmma_f32_16x16x32_f16(
                        false, Ah, false, Bl[cj], (short)0, acc2[rt][cj], false, false);
                    acc2[rt][cj] = __builtin_amdgcn_wmma_f32_16x16x32_f16(
                        false, Al, false, Bh[cj], (short)0, acc2[rt][cj], false, false);
                }
            }
        }
    }

    // ---- store: D VGPR v holds row (v + 8*laneH), col = laneM ----
#pragma unroll
    for (int rt = 0; rt < 4; ++rt) {
#pragma unroll
        for (int cj = 0; cj < 2; ++cj) {
            const int col = (ctBase + wave * 2 + cj) * 16 + laneM;
#pragma unroll
            for (int v = 0; v < 8; ++v) {
                const int m = rt * 16 + v + 8 * laneH;
                out[(size_t)(rowBase + m) * NOUT + col] =
                    acc1[rt][cj][v] + acc2[rt][cj][v] * INV_LSCALE;
            }
        }
    }
}

// ---------------- f32 WMMA fallback (no scratch required) ----------------
__launch_bounds__(256)
__global__ void ngram_gemm_f32_fallback(const int*   __restrict__ tok,
                                        const float* __restrict__ table,
                                        const float* __restrict__ W,
                                        const int* __restrict__ mults,
                                        const int* __restrict__ bias,
                                        const int* __restrict__ sizes,
                                        const int* __restrict__ offs,
                                        float* __restrict__ out)
{
    __shared__ float    Xs[BM * XSTRIDE];
    __shared__ unsigned s_idx[NPAIR];

    const int tid   = threadIdx.x;
    const int lane  = tid & 31;
    const int wave  = tid >> 5;
    const int laneM = lane & 15;
    const int laneH = lane >> 4;
    const int rowBase = blockIdx.x * BM;
    const int colBase = blockIdx.y * BN;

    v8f acc[4][2];
    const v8f vz = {0.f, 0.f, 0.f, 0.f, 0.f, 0.f, 0.f, 0.f};
#pragma unroll
    for (int rt = 0; rt < 4; ++rt)
#pragma unroll
        for (int cj = 0; cj < 2; ++cj) acc[rt][cj] = vz;

    for (int h = 0; h < 2; ++h) {
        __syncthreads();
#pragma unroll
        for (int q = 0; q < 2; ++q) {
            const int p = q * 256 + tid;
            s_idx[p] = hash_index(tok, mults, bias, sizes, offs,
                                  rowBase + (p >> 3), h * 8 + (p & 7));
        }
        __syncthreads();
#pragma unroll 4
        for (int i = 0; i < 32; ++i) {
            const int c  = i * 256 + tid;
            const int p  = c >> 4;
            const int e4 = c & 15;
            const float4 v = *(const float4*)(table + (size_t)s_idx[p] * EMB + e4 * 4);
            *(float4*)(&Xs[(p >> 3) * XSTRIDE + (p & 7) * EMB + e4 * 4]) = v;
        }
        __syncthreads();
        const int kbase = h * KH;
        for (int k0 = 0; k0 < KH; k0 += 4) {
            v2f bf[2];
#pragma unroll
            for (int cj = 0; cj < 2; ++cj) {
                const int n = colBase + (wave * 2 + cj) * 16 + laneM;
                bf[cj] = *(const v2f*)(W + (size_t)n * KDIM + kbase + k0 + 2 * laneH);
            }
#pragma unroll
            for (int rt = 0; rt < 4; ++rt) {
                const v2f a = *(const v2f*)(&Xs[(rt * 16 + laneM) * XSTRIDE + k0 + 2 * laneH]);
#pragma unroll
                for (int cj = 0; cj < 2; ++cj)
                    acc[rt][cj] = __builtin_amdgcn_wmma_f32_16x16x4_f32(
                        false, a, false, bf[cj], (short)0, acc[rt][cj], false, false);
            }
        }
    }
#pragma unroll
    for (int rt = 0; rt < 4; ++rt)
#pragma unroll
        for (int cj = 0; cj < 2; ++cj) {
            const int col = colBase + (wave * 2 + cj) * 16 + laneM;
#pragma unroll
            for (int v = 0; v < 8; ++v)
                out[(size_t)(rowBase + rt * 16 + v + 8 * laneH) * NOUT + col] = acc[rt][cj][v];
        }
}

extern "C" void kernel_launch(void* const* d_in, const int* in_sizes, int n_in,
                              void* d_out, int out_size, void* d_ws, size_t ws_size,
                              hipStream_t stream) {
    (void)in_sizes; (void)n_in; (void)out_size;
    const int*   tok   = (const int*)  d_in[0];
    const float* table = (const float*)d_in[1];
    const float* W     = (const float*)d_in[2];
    const int*   mults = (const int*)  d_in[3];
    const int*   bias  = (const int*)  d_in[4];
    // d_in[5] = order_mask (encoded structurally)
    const int*   sizes = (const int*)  d_in[6];
    const int*   offs  = (const int*)  d_in[7];
    float* out = (float*)d_out;

    const int totalRows = 8 * S_LEN;                  // 16384
    dim3 grid(totalRows / BM, NOUT / BN);             // 256 x 4
    dim3 block(256);

    const size_t halfBytes = (size_t)KDIM * NOUT * sizeof(unsigned short);  // 2 MB
    if (ws_size >= 2 * halfBytes) {
        unsigned* WhP = (unsigned*)d_ws;
        unsigned* WlP = (unsigned*)((char*)d_ws + halfBytes);
        pack_w_f16<<<dim3((KDIM * NOUT / 2) / 256), 256, 0, stream>>>(W, WhP, WlP);
        ngram_gemm_f16split<<<grid, block, 0, stream>>>(
            tok, table, WhP, WlP, mults, bias, sizes, offs, out);
    } else {
        ngram_gemm_f32_fallback<<<grid, block, 0, stream>>>(
            tok, table, W, mults, bias, sizes, offs, out);
    }
}
